// Dilated_res_block_1872605741520
// MI455X (gfx1250) — compile-verified, matrix operand order checked by reading, not run on hardware
//
#include <hip/hip_runtime.h>

#define DEV_INLINE __device__ __forceinline__

typedef __attribute__((ext_vector_type(16))) _Float16 v16h;
typedef __attribute__((ext_vector_type(8)))  float    v8f;

constexpr int  cB = 4;
constexpr int  cN = 40960;
constexpr int  cK = 16;
constexpr long NPTS = (long)cB * cN;   // 163840

// ---- workspace layout (bytes) ----
constexpr size_t FPC_BYTES  = (size_t)NPTS * 32 * sizeof(float);   // f_pc  [B*N,32]
constexpr size_t FAGG_BYTES = (size_t)NPTS * 32 * sizeof(float);   // f_agg [B*N,32]
constexpr size_t WT1_OFF  = FPC_BYTES + FAGG_BYTES;                // wt_fc1h [64][64] f16
constexpr size_t WT2_OFF  = WT1_OFF + 64 * 64 * 2;                 // wt_fc2h [64][64] f16
constexpr size_t WTL_OFF  = WT2_OFF + 64 * 64 * 2;                 // wt_lfa2h [32][32] f16

// ---------------- WMMA fragment helpers ----------------
// 16-bit A (16xK) layout: lane m=lane&15 holds row m; VGPR v holds K pair
// kbase = (v<4?0:16) + (lane>=16?8:0) + 2*(v&3)  -> two contiguous 16B runs.
// B (Kx16) assumed symmetric with column n=lane&15, same K striping; we store
// weights transposed (N-major) so each run is one 16B load.
DEV_INLINE v16h load_frag16(const _Float16* p16) {
  union { float4 f; _Float16 h[8]; } u0, u1;
  u0.f = *reinterpret_cast<const float4*>(p16);
  u1.f = *reinterpret_cast<const float4*>(p16 + 16);
  v16h r;
#pragma unroll
  for (int i = 0; i < 8; ++i) { r[i] = u0.h[i]; r[i + 8] = u1.h[i]; }
  return r;
}
DEV_INLINE v16h load_afrag(const _Float16* fsw, int lane, int koff) {
  int m = lane & 15, g = (lane >> 4) & 1;
  return load_frag16(fsw + m * 64 + koff + g * 8);
}
DEV_INLINE v16h load_bfrag(const _Float16* wt, int kdim, int lane, int ntile, int koff) {
  int n = lane & 15, g = (lane >> 4) & 1;
  return load_frag16(wt + (size_t)(ntile * 16 + n) * kdim + koff + g * 8);
}
DEV_INLINE v8f wmma16(v16h a, v16h b, v8f c) {
  return __builtin_amdgcn_wmma_f32_16x16x32_f16(false, a, false, b, (short)0, c, false, false);
}
DEV_INLINE void wait_lds() { asm volatile("s_wait_dscnt 0" ::: "memory"); }

// ---------------- kernel 0: transpose weights to f16 ----------------
__global__ __launch_bounds__(256) void prep_weights(
    const float* __restrict__ wfc1, const float* __restrict__ wfc2,
    const float* __restrict__ wlfa2,
    _Float16* __restrict__ wt1, _Float16* __restrict__ wt2, _Float16* __restrict__ wtl) {
  int t = threadIdx.x + blockIdx.x * blockDim.x;
  if (t < 64 * 64) {
    int k = t / 64, n = t % 64;
    wt1[n * 64 + k] = (_Float16)wfc1[t];
    wt2[n * 64 + k] = (_Float16)wfc2[t];
  }
  if (t < 32 * 32) {
    int k = t / 32, n = t % 32;
    wtl[n * 32 + k] = (_Float16)wlfa2[t];
  }
}

// ---------------- kernel 1: f_pc = relu(feature@w_mlp1*s+b) ----------------
__global__ __launch_bounds__(256) void mlp1_kernel(
    const float* __restrict__ feature, const float* __restrict__ w,
    const float* __restrict__ s, const float* __restrict__ b,
    float* __restrict__ f_pc) {
  long p = (long)blockIdx.x * 8 + (threadIdx.x >> 5);
  int o = threadIdx.x & 31;
  if (p >= NPTS) return;
  const float* fr = feature + p * 32;
  float acc = 0.f;
#pragma unroll
  for (int i = 0; i < 32; ++i) acc = fmaf(fr[i], w[i * 32 + o], acc);
  f_pc[p * 32 + o] = fmaxf(acc * s[o] + b[o], 0.f);
}

// -------- shared per-point step: rel_pos + lfa1 -> fs[:,32:64] (f16) --------
DEV_INLINE void relpos_lfa1(_Float16* fsw, int lane, long p, long nbp,
                            const float* __restrict__ xyz,
                            const float* __restrict__ w_lfa1,
                            const float* __restrict__ s_lfa1,
                            const float* __restrict__ b_lfa1) {
  int k = lane & 15, cb = lane >> 4;
  const float* xc = xyz + p * 3;
  const float* xn = xyz + nbp * 3;
  float tx = xc[0], ty = xc[1], tz = xc[2];
  float nx = xn[0], ny = xn[1], nz = xn[2];
  float rx = tx - nx, ry = ty - ny, rz = tz - nz;
  float dist = sqrtf(rx * rx + ry * ry + rz * rz);
  float in10[10] = {dist, rx, ry, rz, tx, ty, tz, nx, ny, nz};
#pragma unroll
  for (int oo = 0; oo < 16; ++oo) {
    int o = cb * 16 + oo;
    float acc = 0.f;
#pragma unroll
    for (int i = 0; i < 10; ++i) acc = fmaf(in10[i], w_lfa1[i * 32 + o], acc);
    acc = fmaxf(acc * s_lfa1[o] + b_lfa1[o], 0.f);
    fsw[k * 64 + 32 + o] = (_Float16)acc;
  }
}

// -------- shared: WMMA 16x64 @ 64x64, column softmax over K=16, agg --------
DEV_INLINE void att_core(const _Float16* fsw, float* aggw, int lane,
                         const _Float16* __restrict__ wt_fc) {
  v16h a0 = load_afrag(fsw, lane, 0);
  v16h a1 = load_afrag(fsw, lane, 32);
  int g = lane >> 4, nn = lane & 15;
#pragma unroll
  for (int j = 0; j < 4; ++j) {
    v8f c = {};
    c = wmma16(a0, load_bfrag(wt_fc, 64, lane, j, 0), c);
    c = wmma16(a1, load_bfrag(wt_fc, 64, lane, j, 32), c);
    // softmax over the 16 neighbor rows of this 16-col tile
    float mx = c[0];
#pragma unroll
    for (int v = 1; v < 8; ++v) mx = fmaxf(mx, c[v]);
    mx = fmaxf(mx, __shfl_xor(mx, 16, 32));
    float e[8], sum = 0.f;
#pragma unroll
    for (int v = 0; v < 8; ++v) { e[v] = __expf(c[v] - mx); sum += e[v]; }
    sum += __shfl_xor(sum, 16, 32);
    float inv = 1.f / sum;
    int cc = j * 16 + nn;
    float acc = 0.f;
#pragma unroll
    for (int v = 0; v < 8; ++v)
      acc = fmaf((float)fsw[(v + 8 * g) * 64 + cc], e[v], acc);
    acc *= inv;
    acc += __shfl_xor(acc, 16, 32);
    if (g == 0) aggw[cc] = acc;
  }
}

// ---------------- kernel 2: LFA1 + attention pool 1 -> f_agg ----------------
__global__ __launch_bounds__(256) void att_block1(
    const float* __restrict__ xyz, const int* __restrict__ nidx,
    const float* __restrict__ f_pc,
    const float* __restrict__ w_lfa1, const float* __restrict__ s_lfa1,
    const float* __restrict__ b_lfa1,
    const _Float16* __restrict__ wt_fc1,
    const float* __restrict__ w_mlp, const float* __restrict__ s_mlp,
    const float* __restrict__ b_mlp,
    float* __restrict__ f_agg) {
  __shared__ _Float16 fs[8][16 * 64];
  __shared__ float agg[8][64];
  int w = threadIdx.x >> 5, lane = threadIdx.x & 31;
  long p = (long)blockIdx.x * 8 + w;
  if (p >= NPTS) p = NPTS - 1;
  int bidx = (int)(p / cN);
  _Float16* fsw = fs[w];
  float* aggw = agg[w];

  int k = lane & 15, h = lane >> 4;
  int nb = nidx[p * cK + k];
  long nbp = (long)bidx * cN + nb;

  relpos_lfa1(fsw, lane, p, nbp, xyz, w_lfa1, s_lfa1, b_lfa1);   // fs[:,32:64]
  {                                                              // fs[:,0:32] = gather f_pc
    const float* src = f_pc + nbp * 32 + h * 16;
#pragma unroll
    for (int i = 0; i < 16; ++i) fsw[k * 64 + h * 16 + i] = (_Float16)src[i];
  }
  wait_lds();
  att_core(fsw, aggw, lane, wt_fc1);
  wait_lds();
  {                                                              // f_agg = relu(agg@W64x32*s+b)
    int o = lane;
    float acc = 0.f;
#pragma unroll
    for (int c2 = 0; c2 < 64; ++c2) acc = fmaf(aggw[c2], w_mlp[c2 * 32 + o], acc);
    f_agg[p * 32 + o] = fmaxf(acc * s_mlp[o] + b_mlp[o], 0.f);
  }
}

// --- kernel 3: recompute LFA1, LFA2 (WMMA), attention 2, mlp2+shortcut ---
__global__ __launch_bounds__(256) void att_block2(
    const float* __restrict__ xyz, const int* __restrict__ nidx,
    const float* __restrict__ feature, const float* __restrict__ f_agg,
    const float* __restrict__ w_lfa1, const float* __restrict__ s_lfa1,
    const float* __restrict__ b_lfa1,
    const _Float16* __restrict__ wt_lfa2,
    const float* __restrict__ s_lfa2, const float* __restrict__ b_lfa2,
    const _Float16* __restrict__ wt_fc2,
    const float* __restrict__ w_amlp, const float* __restrict__ s_amlp,
    const float* __restrict__ b_amlp,
    const float* __restrict__ w_mlp2, const float* __restrict__ s_mlp2,
    const float* __restrict__ b_mlp2,
    const float* __restrict__ w_sc, const float* __restrict__ s_sc,
    const float* __restrict__ b_sc,
    float* __restrict__ out) {
  __shared__ _Float16 fs[8][16 * 64];
  __shared__ float agg[8][64];
  int w = threadIdx.x >> 5, lane = threadIdx.x & 31;
  long p = (long)blockIdx.x * 8 + w;
  if (p >= NPTS) p = NPTS - 1;
  int bidx = (int)(p / cN);
  _Float16* fsw = fs[w];
  float* aggw = agg[w];
  int g = lane >> 4, nn = lane & 15;

  int k = lane & 15, h = lane >> 4;
  int nb = nidx[p * cK + k];
  long nbp = (long)bidx * cN + nb;

  relpos_lfa1(fsw, lane, p, nbp, xyz, w_lfa1, s_lfa1, b_lfa1);   // fs[:,32:64] = f_xyz
  wait_lds();
  {                               // f_xyz2 = relu(f_xyz @ w_lfa2 * s + b) via WMMA
    v16h ax = load_afrag(fsw, lane, 32);
    v8f y0 = {}, y1 = {};
    y0 = wmma16(ax, load_bfrag(wt_lfa2, 32, lane, 0, 0), y0);
    y1 = wmma16(ax, load_bfrag(wt_lfa2, 32, lane, 1, 0), y1);
#pragma unroll
    for (int j = 0; j < 2; ++j) {
      v8f& y = j ? y1 : y0;
      int c = j * 16 + nn;
      float sv = s_lfa2[c], bv = b_lfa2[c];
#pragma unroll
      for (int v = 0; v < 8; ++v)
        fsw[(v + 8 * g) * 64 + 32 + c] = (_Float16)fmaxf(y[v] * sv + bv, 0.f);
    }
  }
  {                               // fs[:,0:32] = gather f_agg
    const float* src = f_agg + nbp * 32 + h * 16;
#pragma unroll
    for (int i = 0; i < 16; ++i) fsw[k * 64 + h * 16 + i] = (_Float16)src[i];
  }
  wait_lds();
  att_core(fsw, aggw, lane, wt_fc2);
  wait_lds();

  float* pc2 = reinterpret_cast<float*>(fsw);   // fs region reused as f32 scratch
  float* feat = pc2 + 64;
#pragma unroll
  for (int q = 0; q < 2; ++q) {   // f_pc2 = relu(agg @ W64x64 * s + b)
    int o = lane + q * 32;
    float acc = 0.f;
#pragma unroll
    for (int c2 = 0; c2 < 64; ++c2) acc = fmaf(aggw[c2], w_amlp[c2 * 64 + o], acc);
    pc2[o] = fmaxf(acc * s_amlp[o] + b_amlp[o], 0.f);
  }
  feat[lane] = feature[p * 32 + lane];
  wait_lds();
#pragma unroll
  for (int q = 0; q < 4; ++q) {   // out = leaky_relu(pc2@W64x128 + feat@Wsc)
    int o = lane + q * 32;
    float acc = 0.f;
#pragma unroll
    for (int c2 = 0; c2 < 64; ++c2) acc = fmaf(pc2[c2], w_mlp2[c2 * 128 + o], acc);
    acc = acc * s_mlp2[o] + b_mlp2[o];
    float sc = 0.f;
#pragma unroll
    for (int i = 0; i < 32; ++i) sc = fmaf(feat[i], w_sc[i * 128 + o], sc);
    sc = sc * s_sc[o] + b_sc[o];
    float t = acc + sc;
    out[p * 128 + o] = t > 0.f ? t : 0.2f * t;
  }
}

extern "C" void kernel_launch(void* const* d_in, const int* in_sizes, int n_in,
                              void* d_out, int out_size, void* d_ws, size_t ws_size,
                              hipStream_t stream) {
  const float* feature = (const float*)d_in[0];
  const float* xyz     = (const float*)d_in[1];
  const int*   nidx    = (const int*)d_in[2];
  const float* w_mlp1  = (const float*)d_in[3];
  const float* s_mlp1  = (const float*)d_in[4];
  const float* b_mlp1  = (const float*)d_in[5];
  const float* w_lfa1  = (const float*)d_in[6];
  const float* s_lfa1  = (const float*)d_in[7];
  const float* b_lfa1  = (const float*)d_in[8];
  const float* w_a1fc  = (const float*)d_in[9];
  const float* w_a1m   = (const float*)d_in[10];
  const float* s_a1m   = (const float*)d_in[11];
  const float* b_a1m   = (const float*)d_in[12];
  const float* w_lfa2  = (const float*)d_in[13];
  const float* s_lfa2  = (const float*)d_in[14];
  const float* b_lfa2  = (const float*)d_in[15];
  const float* w_a2fc  = (const float*)d_in[16];
  const float* w_a2m   = (const float*)d_in[17];
  const float* s_a2m   = (const float*)d_in[18];
  const float* b_a2m   = (const float*)d_in[19];
  const float* w_mlp2  = (const float*)d_in[20];
  const float* s_mlp2  = (const float*)d_in[21];
  const float* b_mlp2  = (const float*)d_in[22];
  const float* w_sc    = (const float*)d_in[23];
  const float* s_sc    = (const float*)d_in[24];
  const float* b_sc    = (const float*)d_in[25];
  float* out = (float*)d_out;

  char* ws = (char*)d_ws;
  float*    f_pc   = (float*)ws;
  float*    f_agg  = (float*)(ws + FPC_BYTES);
  _Float16* wt_fc1 = (_Float16*)(ws + WT1_OFF);
  _Float16* wt_fc2 = (_Float16*)(ws + WT2_OFF);
  _Float16* wt_l2  = (_Float16*)(ws + WTL_OFF);

  prep_weights<<<16, 256, 0, stream>>>(w_a1fc, w_a2fc, w_lfa2, wt_fc1, wt_fc2, wt_l2);

  int blocks = (int)((NPTS + 7) / 8);
  mlp1_kernel<<<blocks, 256, 0, stream>>>(feature, w_mlp1, s_mlp1, b_mlp1, f_pc);

  att_block1<<<blocks, 256, 0, stream>>>(xyz, nidx, f_pc,
                                         w_lfa1, s_lfa1, b_lfa1,
                                         wt_fc1, w_a1m, s_a1m, b_a1m, f_agg);

  att_block2<<<blocks, 256, 0, stream>>>(xyz, nidx, feature, f_agg,
                                         w_lfa1, s_lfa1, b_lfa1,
                                         wt_l2, s_lfa2, b_lfa2,
                                         wt_fc2, w_a2m, s_a2m, b_a2m,
                                         w_mlp2, s_mlp2, b_mlp2,
                                         w_sc, s_sc, b_sc, out);
}
